// MoEGate_9835475107966
// MI455X (gfx1250) — matrix operand order repaired and checked by reading, not compiled, not run on hardware
//
#include <hip/hip_runtime.h>
#include <hip/hip_bf16.h>
#include <math.h>

#define TOKENS 16384
#define KDIM   2048
#define NEXP   64
#define TOPK   6
#define KSTEP  32
#define NCHUNK (KDIM / KSTEP)      /* 64 K-chunks */
#define CHUNK_DW 1024              /* 4 N-tiles * 32 lanes * 8 dwords per chunk */
#define ROWS_PER_BLOCK 128         /* 8 waves * 16 rows */

typedef __attribute__((ext_vector_type(16))) __bf16 v16bf;
typedef __attribute__((ext_vector_type(8)))  float  v8f;

// round-to-nearest-even fp32 -> bf16 (as uint16 in low bits)
__device__ __forceinline__ unsigned f2bf_rne(float f) {
    unsigned u = __float_as_uint(f);
    return (u + 0x7FFFu + ((u >> 16) & 1u)) >> 16;
}

// split two floats into packed bf16 hi pair (returned) and bf16 lo pair (out param)
__device__ __forceinline__ unsigned cvt_pair(float a, float b, unsigned& lo) {
    unsigned ha = f2bf_rne(a), hb = f2bf_rne(b);
    float ra = a - __uint_as_float(ha << 16);
    float rb = b - __uint_as_float(hb << 16);
    lo = f2bf_rne(ra) | (f2bf_rne(rb) << 16);
    return ha | (hb << 16);
}

// ---------------------------------------------------------------------------
// Prologue: reorder W_g [2048,64] fp32 into WMMA B-operand layout, bf16 hi/lo.
// B frag as packed here, contiguous K matching A (8 dwords/lane): lanes 0-15 hold K=0..15 (2/VGPR),
// lanes 16-31 hold K=16..31; lane%16 = column N within the 16-wide tile.
// ws dword index = ((c*4 + t)*32 + lane)*8 + j
// ---------------------------------------------------------------------------
__global__ void moe_prep_B(const float* __restrict__ W,
                           unsigned* __restrict__ bhi,
                           unsigned* __restrict__ blo) {
    int idx  = blockIdx.x * blockDim.x + threadIdx.x;   // 0 .. 65535
    int j    = idx & 7;
    int lane = (idx >> 3) & 31;
    int t    = (idx >> 8) & 3;
    int c    = idx >> 10;
    int n    = t * 16 + (lane & 15);
    int k0   = c * KSTEP + ((lane >> 4) << 4) + 2 * j;  // even K of the pair
    float w0 = W[(size_t)k0 * NEXP + n];
    float w1 = W[(size_t)(k0 + 1) * NEXP + n];
    unsigned lo0, lo1;
    unsigned hi0 = cvt_pair(w0, w1, lo0);
    (void)lo1;
    bhi[idx] = hi0;
    blo[idx] = lo0;
}

// ---------------------------------------------------------------------------
// Main fused kernel: GEMM (bf16-split WMMA) + softmax + top-6.
// ---------------------------------------------------------------------------
__global__ void __launch_bounds__(256)
moe_gate(const float* __restrict__ x,
         const unsigned* __restrict__ bhi,
         const unsigned* __restrict__ blo,
         int* __restrict__ oidx,
         float* __restrict__ oscr) {
    __shared__ unsigned ldsB[2][2 * CHUNK_DW];   // double-buffered: [hi 4KB | lo 4KB]

    const int tid  = threadIdx.x;
    const int lane = tid & 31;
    const int wave = tid >> 5;
    const int lr   = lane & 15;   // column-in-tile / row-in-tile selector
    const int hs   = lane >> 4;   // half-wave select
    const int rowBase = blockIdx.x * ROWS_PER_BLOCK + wave * 16;

    // Stage B chunk 0 into buffer 0 (each thread moves 4 hi + 4 lo dwords)
    {
        uint4 h = *(const uint4*)(bhi + tid * 4);
        uint4 l = *(const uint4*)(blo + tid * 4);
        *(uint4*)(&ldsB[0][tid * 4])            = h;
        *(uint4*)(&ldsB[0][CHUNK_DW + tid * 4]) = l;
    }

    union CFrag { v8f v; float f[8]; };
    CFrag acc[4];
#pragma unroll
    for (int t = 0; t < 4; ++t)
#pragma unroll
        for (int e = 0; e < 8; ++e) acc[t].f[e] = 0.0f;

    // A operand rows: lane L and L+16 both serve row rowBase + (L&15)
    const float* xrow = x + (size_t)(rowBase + lr) * KDIM;

    for (int c = 0; c < NCHUNK; ++c) {
        const bool more = (c + 1 < NCHUNK);
        uint4 nh = {}, nl = {};
        if (more) {  // prefetch next B chunk from L2 into registers (pre-barrier)
            nh = *(const uint4*)(bhi + (size_t)(c + 1) * CHUNK_DW + tid * 4);
            nl = *(const uint4*)(blo + (size_t)(c + 1) * CHUNK_DW + tid * 4);
        }
        __syncthreads();  // buffer c&1 now visible; prior reads of buffer (c+1)&1 retired

        // ---- A: 16 fp32 per lane, contiguous K = 16*hs + s (same order as B), split hi/lo
        const int kb = c * KSTEP + hs * 16;
        float4 f0 = *(const float4*)(xrow + kb);
        float4 f1 = *(const float4*)(xrow + kb + 4);
        float4 f2 = *(const float4*)(xrow + kb + 8);
        float4 f3 = *(const float4*)(xrow + kb + 12);

        union AFrag { unsigned u[8]; v16bf v; } ahi, alo;
        ahi.u[0] = cvt_pair(f0.x, f0.y, alo.u[0]);
        ahi.u[1] = cvt_pair(f0.z, f0.w, alo.u[1]);
        ahi.u[2] = cvt_pair(f1.x, f1.y, alo.u[2]);
        ahi.u[3] = cvt_pair(f1.z, f1.w, alo.u[3]);
        ahi.u[4] = cvt_pair(f2.x, f2.y, alo.u[4]);
        ahi.u[5] = cvt_pair(f2.z, f2.w, alo.u[5]);
        ahi.u[6] = cvt_pair(f3.x, f3.y, alo.u[6]);
        ahi.u[7] = cvt_pair(f3.z, f3.w, alo.u[7]);

        // ---- B from LDS, 3-product split accumulation per N-tile
        const unsigned* buf = &ldsB[c & 1][0];
#pragma unroll
        for (int t = 0; t < 4; ++t) {
            union BFrag { uint4 q[2]; v16bf v; } bh, bl;
            const int bb = t * 256 + lane * 8;
            bh.q[0] = *(const uint4*)(buf + bb);
            bh.q[1] = *(const uint4*)(buf + bb + 4);
            bl.q[0] = *(const uint4*)(buf + CHUNK_DW + bb);
            bl.q[1] = *(const uint4*)(buf + CHUNK_DW + bb + 4);
            acc[t].v = __builtin_amdgcn_wmma_f32_16x16x32_bf16(
                false, ahi.v, false, bh.v, (short)0, acc[t].v, false, false);
            acc[t].v = __builtin_amdgcn_wmma_f32_16x16x32_bf16(
                false, ahi.v, false, bl.v, (short)0, acc[t].v, false, false);
            acc[t].v = __builtin_amdgcn_wmma_f32_16x16x32_bf16(
                false, alo.v, false, bh.v, (short)0, acc[t].v, false, false);
        }

        if (more) {  // stage prefetched chunk into the other buffer
            unsigned* w = &ldsB[(c + 1) & 1][0];
            *(uint4*)(w + tid * 4)            = nh;
            *(uint4*)(w + CHUNK_DW + tid * 4) = nl;
        }
    }

    // ------------------------------------------------------------------
    // Softmax + top-6. C layout: VGPR r, lanes 0-15 -> M=r, lanes 16-31 ->
    // M=r+8; column = t*16 + (lane&15). Each row lives in one 16-lane group,
    // 4 values per lane. All reductions via xor-shuffles with mask < 16.
    // ------------------------------------------------------------------
#pragma unroll 1
    for (int r = 0; r < 8; ++r) {
        const int row = rowBase + r + 8 * hs;
        float v0 = acc[0].f[r], v1 = acc[1].f[r], v2 = acc[2].f[r], v3 = acc[3].f[r];

        float m = fmaxf(fmaxf(v0, v1), fmaxf(v2, v3));
        for (int off = 8; off >= 1; off >>= 1) m = fmaxf(m, __shfl_xor(m, off));

        float s = __expf(v0 - m) + __expf(v1 - m) + __expf(v2 - m) + __expf(v3 - m);
        for (int off = 8; off >= 1; off >>= 1) s += __shfl_xor(s, off);

        float w0 = v0, w1 = v1, w2 = v2, w3 = v3;
        for (int i = 0; i < TOPK; ++i) {
            float bv = w0; int bt = 0;
            if (w1 > bv) { bv = w1; bt = 1; }
            if (w2 > bv) { bv = w2; bt = 2; }
            if (w3 > bv) { bv = w3; bt = 3; }
            int bc = bt * 16 + lr;
            for (int off = 8; off >= 1; off >>= 1) {
                float ov = __shfl_xor(bv, off);
                int   oc = __shfl_xor(bc, off);
                if (ov > bv || (ov == bv && oc < bc)) { bv = ov; bc = oc; }
            }
            if ((bc & 15) == lr) {          // winning lane removes its value
                int t = bc >> 4;
                if      (t == 0) w0 = -INFINITY;
                else if (t == 1) w1 = -INFINITY;
                else if (t == 2) w2 = -INFINITY;
                else             w3 = -INFINITY;
            }
            if (lr == 0) {
                oidx[(size_t)row * TOPK + i] = bc;
                oscr[(size_t)row * TOPK + i] = __expf(bv - m) / s;
            }
        }
    }
}

// ---------------------------------------------------------------------------
extern "C" void kernel_launch(void* const* d_in, const int* in_sizes, int n_in,
                              void* d_out, int out_size, void* d_ws, size_t ws_size,
                              hipStream_t stream) {
    (void)in_sizes; (void)n_in; (void)out_size; (void)ws_size;
    const float* x = (const float*)d_in[0];
    const float* W = (const float*)d_in[1];

    // Workspace: bf16-hi B operand (256 KB) then bf16-lo B operand (256 KB)
    unsigned* bhi = (unsigned*)d_ws;
    unsigned* blo = bhi + (size_t)NCHUNK * CHUNK_DW;

    // Output: [tokens*6] int32 indices, then [tokens*6] float32 scores
    int*   oidx = (int*)d_out;
    float* oscr = (float*)d_out + (size_t)TOKENS * TOPK;

    moe_prep_B<<<(NCHUNK * CHUNK_DW) / 256, 256, 0, stream>>>(W, bhi, blo);
    moe_gate<<<TOKENS / ROWS_PER_BLOCK, 256, 0, stream>>>(x, bhi, blo, oidx, oscr);
}
